// GCNWithMultiHeadGATAndTCN_42356967473538
// MI455X (gfx1250) — matrix operand order repaired and checked by reading, not compiled, not run on hardware
//
#include <hip/hip_runtime.h>
#include <stdint.h>

// MI455X / gfx1250: bf16 WMMA (16x16x32) everywhere, TDM (tensor_load_to_lds)
// double-buffered staging of K-contiguous (transposed) B tiles into LDS,
// flash-softmax attention (no NxN materialization), conv1d as a K=768 WMMA
// GEMM with implicit zero padding, training-mode batchnorm.
// All matrix fragments are two 16-byte loads per lane.

typedef __bf16 bf16_t;
typedef __attribute__((ext_vector_type(16))) __bf16   v16bf;
typedef __attribute__((ext_vector_type(8)))  float    v8f;
typedef __attribute__((ext_vector_type(4)))  uint32_t u32x4;
typedef __attribute__((ext_vector_type(8)))  uint32_t u32x8;

union AFrag { v16bf v; u32x4 q[2]; };
union BFrag { v16bf v; u32x4 q[2]; };

#define WMMA_BF16(A, B, C) \
  __builtin_amdgcn_wmma_f32_16x16x32_bf16(false, (A), false, (B), (short)0, (C), false, false)

// ---------------------------------------------------------------- TDM issue
// 2-D D#: tile of 64 rows x 32 bf16 (a K-contiguous B tile), row stride =
// ldbt elements, DMA'd row-major into LDS at laddr (-> LDS layout [64][32]).
__device__ __forceinline__ void tdm_load_bt_tile(uint32_t laddr, const bf16_t* gsrc,
                                                 int ldbt) {
  uint64_t ga = (uint64_t)(uintptr_t)gsrc;
  u32x4 g0;
  g0[0] = 1u;                                   // count=1, user descriptor
  g0[1] = laddr;                                // lds_addr (bytes)
  g0[2] = (uint32_t)ga;                         // global_addr[31:0]
  g0[3] = (uint32_t)((ga >> 32) & 0x1FFFFFFu)   // global_addr[56:32]
          | (2u << 30);                         // type = 2 ("image")
  u32x8 g1;
  g1[0] = (1u << 16);                           // workgroup_mask=0, data_size=1 (2B)
  g1[1] = (32u << 16);                          // tensor_dim0[15:0] = 32
  g1[2] = (64u << 16);                          // tensor_dim1[15:0] = 64
  g1[3] = (32u << 16);                          // tile_dim0 = 32
  g1[4] = 64u;                                  // tile_dim1 = 64, tile_dim2 = 0
  g1[5] = (uint32_t)ldbt;                       // tensor_dim0_stride[31:0]
  g1[6] = 0u;
  g1[7] = 0u;
  asm volatile("tensor_load_to_lds %0, %1" :: "s"(g0), "s"(g1) : "memory");
}

// ---------------------------------------------------------------- converts
__global__ void f2bf_kernel(const float* __restrict__ in, bf16_t* __restrict__ out, long n) {
  long i = (long)blockIdx.x * blockDim.x + threadIdx.x;
  if (i < n) out[i] = (bf16_t)in[i];
}

// W_sage [k=256][n=256] -> WsT[n][k]
__global__ void ws_transpose_kernel(const float* __restrict__ w, bf16_t* __restrict__ wT) {
  long i = (long)blockIdx.x * blockDim.x + threadIdx.x;
  if (i < 65536) {
    int k = (int)(i & 255), n = (int)(i >> 8);
    wT[i] = (bf16_t)w[k * 256 + n];
  }
}

// Wh [h][k=256][n=64] -> WhT[h][n][k]
__global__ void wh_transpose_kernel(const float* __restrict__ wh, bf16_t* __restrict__ whT) {
  long i = (long)blockIdx.x * blockDim.x + threadIdx.x;
  if (i < 65536) {
    int k = (int)(i & 255), n = (int)((i >> 8) & 63), h = (int)(i >> 14);
    whT[i] = (bf16_t)wh[((long)h * 256 + k) * 64 + n];
  }
}

// conv_w (OIH: [o=256][i=256][tap=3]) -> wpackT[o][kk=tap*256+i]  ([256,768])
__global__ void conv_pack_kernel(const float* __restrict__ cw, bf16_t* __restrict__ wT) {
  long i = (long)blockIdx.x * blockDim.x + threadIdx.x;
  if (i < 256l * 768) {
    int kk = (int)(i % 768), o = (int)(i / 768);
    int tap = kk >> 8, ic = kk & 255;
    wT[i] = (bf16_t)cw[((long)o * 256 + ic) * 3 + tap];
  }
}

// hp [b][node][h*64+d] -> hpT [b][h][d][node]
__global__ void hp_transpose_kernel(const bf16_t* __restrict__ hp, bf16_t* __restrict__ hpT) {
  long i = (long)blockIdx.x * blockDim.x + threadIdx.x;
  if (i < 4l * 4 * 64 * 2048) {
    int node = (int)(i & 2047);
    int d    = (int)((i >> 11) & 63);
    int h    = (int)((i >> 17) & 3);
    int b    = (int)(i >> 19);
    hpT[i] = hp[((long)b * 2048 + node) * 256 + h * 64 + d];
  }
}

// ------------------------------------------------------------ staged GEMM
// Block = 256 threads (8 waves). Block tile M=128, N=64. BT is K-contiguous
// ([n][k]). Wave 0 DMAs the shared 64x32 BT tile into LDS (double-buffered,
// overlapped with compute); every wave reuses its A fragment over 4 N-tiles.
template <int RELU, bool OUTF, bool OUTH, bool OUTT>
__global__ __launch_bounds__(256)
void gemm_tdm_kernel(const bf16_t* __restrict__ A, const bf16_t* __restrict__ BT,
                     const float* __restrict__ bias,
                     float* __restrict__ Cf, bf16_t* __restrict__ Ch,
                     bf16_t* __restrict__ Ct,
                     int M, int N, int K, int lda, int ldbt, int ldc, int ldct,
                     long sA, long sB, long sC, long sCt) {
  __shared__ bf16_t btile[2][64 * 32];
  int z = blockIdx.z;
  A  += (long)z * sA;
  BT += (long)z * sB;
  long coff  = (long)z * sC;
  long cofft = (long)z * sCt;

  int wave = threadIdx.x >> 5, lane = threadIdx.x & 31;
  int g = lane >> 4, r = lane & 15;
  int tm  = blockIdx.y * 8 + wave;
  int bn0 = blockIdx.x * 64;

  const bf16_t* bsrc = BT + (long)bn0 * ldbt;
  const bf16_t* arow = A + (long)(tm * 16 + r) * lda;

  v8f zero = {};
  v8f acc[4] = {zero, zero, zero, zero};
  int nk = K >> 5;

  if (wave == 0)
    tdm_load_bt_tile((uint32_t)(uintptr_t)(void*)btile[0], bsrc, ldbt);

  for (int i = 0; i < nk; ++i) {
    if (wave == 0) __builtin_amdgcn_s_wait_tensorcnt(0);
    __syncthreads();
    if (wave == 0 && i + 1 < nk)
      tdm_load_bt_tile((uint32_t)(uintptr_t)(void*)btile[(i + 1) & 1],
                       bsrc + (i + 1) * 32, ldbt);
    const bf16_t* bt = btile[i & 1];
    int k0 = i << 5;
    __builtin_prefetch((const void*)(arow + k0 + 64), 0, 1);  // global_prefetch_b8
    AFrag a;                                     // A 16x32: two contiguous 16B runs
    a.q[0] = *(const u32x4*)(arow + k0 + g * 16);
    a.q[1] = *(const u32x4*)(arow + k0 + g * 16 + 8);
#pragma unroll
    for (int c = 0; c < 4; ++c) {
      BFrag b;                                   // lane = col, 16 contiguous K
      const u32x4* lp = (const u32x4*)(bt + (c * 16 + r) * 32 + g * 16);
      b.q[0] = lp[0];
      b.q[1] = lp[1];
      acc[c] = WMMA_BF16(a.v, b.v, acc[c]);
    }
  }

#pragma unroll
  for (int c = 0; c < 4; ++c) {
    int col = bn0 + c * 16 + r;
    float bs = bias ? bias[col] : 0.0f;
    float vals[8];
#pragma unroll
    for (int v = 0; v < 8; ++v) {
      float x = acc[c][v] + bs;
      if (RELU) x = fmaxf(x, 0.0f);
      vals[v] = x;
    }
    if (OUTF || OUTH) {
#pragma unroll
      for (int v = 0; v < 8; ++v) {
        int row = tm * 16 + v + 8 * g;
        long idx = coff + (long)row * ldc + col;
        if (OUTF) Cf[idx] = vals[v];
        if (OUTH) Ch[idx] = (bf16_t)vals[v];
      }
    }
    if (OUTT) {                                  // transposed: 8 rows contiguous
      union { u32x4 q; bf16_t b[8]; } pk;
#pragma unroll
      for (int v = 0; v < 8; ++v) pk.b[v] = (bf16_t)vals[v];
      *(u32x4*)(Ct + cofft + (long)col * ldct + tm * 16 + 8 * g) = pk.q;
    }
  }
}

// ------------------------------------------------- conv1d (k=3) as WMMA GEMM
// T[b,n,o] = sum_{tap,i} G[b, n+tap-1, i] * conv_w[o,i,tap] + bias[o]
// B = wpackT [o][768] K-contiguous, TDM-staged; A rows shifted per tap.
__global__ __launch_bounds__(256)
void wmma_conv_kernel(const bf16_t* __restrict__ G, const bf16_t* __restrict__ WT,
                      const float* __restrict__ bias, float* __restrict__ T) {
  __shared__ bf16_t btile[2][64 * 32];
  int z = blockIdx.z;
  const bf16_t* A = G + (long)z * 2048 * 256;
  int wave = threadIdx.x >> 5, lane = threadIdx.x & 31;
  int g = lane >> 4, r = lane & 15;
  int tm  = blockIdx.y * 8 + wave;
  int bn0 = blockIdx.x * 64;
  int n = tm * 16 + r;

  const bf16_t* bsrc = WT + (long)bn0 * 768;

  v8f zero = {};
  v8f acc[4] = {zero, zero, zero, zero};

  if (wave == 0)
    tdm_load_bt_tile((uint32_t)(uintptr_t)(void*)btile[0], bsrc, 768);

  for (int i = 0; i < 24; ++i) {
    if (wave == 0) __builtin_amdgcn_s_wait_tensorcnt(0);
    __syncthreads();
    if (wave == 0 && i + 1 < 24)
      tdm_load_bt_tile((uint32_t)(uintptr_t)(void*)btile[(i + 1) & 1],
                       bsrc + (i + 1) * 32, 768);
    const bf16_t* bt = btile[i & 1];
    int k0 = i << 5;
    int tap = k0 >> 8;                          // chunk never crosses a tap
    int ar  = n + tap - 1;
    bool ok = (unsigned)ar < 2048u;             // implicit zero padding
    int kb  = k0 & 255;
    AFrag a;
    if (ok) {
      a.q[0] = *(const u32x4*)(A + (long)ar * 256 + kb + g * 16);
      a.q[1] = *(const u32x4*)(A + (long)ar * 256 + kb + g * 16 + 8);
    } else {
      u32x4 zq = {};
      a.q[0] = zq; a.q[1] = zq;
    }
#pragma unroll
    for (int c = 0; c < 4; ++c) {
      BFrag b;
      const u32x4* lp = (const u32x4*)(bt + (c * 16 + r) * 32 + g * 16);
      b.q[0] = lp[0];
      b.q[1] = lp[1];
      acc[c] = WMMA_BF16(a.v, b.v, acc[c]);
    }
  }

#pragma unroll
  for (int c = 0; c < 4; ++c) {
    int col = bn0 + c * 16 + r;
    float bs = bias[col];
#pragma unroll
    for (int v = 0; v < 8; ++v)
      T[(long)z * 2048 * 256 + (long)(tm * 16 + v + 8 * g) * 256 + col] = acc[c][v] + bs;
  }
}

// --------------------------------------------------------------- batchnorm
__global__ __launch_bounds__(256)
void bn_stats_kernel(const float* __restrict__ x, int rows, int C,
                     float* __restrict__ mean, float* __restrict__ rstd) {
  int c = blockIdx.x, tid = threadIdx.x;
  float s = 0.f, s2 = 0.f;
  for (int r = tid; r < rows; r += blockDim.x) {
    float v = x[(long)r * C + c];
    s += v; s2 += v * v;
  }
  __shared__ float sh[256], sh2[256];
  sh[tid] = s; sh2[tid] = s2; __syncthreads();
  for (int off = 128; off > 0; off >>= 1) {
    if (tid < off) { sh[tid] += sh[tid + off]; sh2[tid] += sh2[tid + off]; }
    __syncthreads();
  }
  if (tid == 0) {
    float m = sh[0] / rows;
    float var = sh2[0] / rows - m * m;           // biased (training-mode)
    mean[c] = m;
    rstd[c] = rsqrtf(var + 1e-5f);
  }
}

__global__ void bn_apply_relu_kernel(const float* __restrict__ x, const float* __restrict__ mean,
                                     const float* __restrict__ rstd, const float* __restrict__ gamma,
                                     const float* __restrict__ beta,
                                     float* __restrict__ yf, bf16_t* __restrict__ yh, long n, int C) {
  long i = (long)blockIdx.x * blockDim.x + threadIdx.x;
  if (i < n) {
    int c = (int)(i % C);
    float y = gamma[c] * (x[i] - mean[c]) * rstd[c] + beta[c];
    y = fmaxf(y, 0.0f);
    yf[i] = y;
    yh[i] = (bf16_t)y;
  }
}

__global__ void mix_kernel(const float* __restrict__ a_in, const float* __restrict__ b_in,
                           const float* __restrict__ alpha,
                           float* __restrict__ yf, bf16_t* __restrict__ yh, long n) {
  long i = (long)blockIdx.x * blockDim.x + threadIdx.x;
  if (i < n) {
    float a = *alpha;
    float v = a * a_in[i] + (1.0f - a) * b_in[i];
    yf[i] = v;
    yh[i] = (bf16_t)v;
  }
}

__global__ void final_kernel(const float* __restrict__ t, const float* __restrict__ mean,
                             const float* __restrict__ rstd, const float* __restrict__ gamma,
                             const float* __restrict__ beta, const float* __restrict__ alpha,
                             const float* __restrict__ gres, float* __restrict__ out, long n, int C) {
  long i = (long)blockIdx.x * blockDim.x + threadIdx.x;
  if (i < n) {
    int c = (int)(i % C);
    float a = *alpha;
    float bn = gamma[c] * (t[i] - mean[c]) * rstd[c] + beta[c];
    bn = fmaxf(bn, 0.0f);
    out[i] = a * bn + (1.0f - a) * gres[i];
  }
}

// ------------------------------------------------------- flash GAT attention
// One wave == one 16-row query tile for one (b, h). Streams 32 keys/iter:
// S = Q Kt via WMMA, LeakyReLU, online softmax (register max/sum), P tile
// through per-wave LDS to reshape into an A-fragment, then acc += P V (WMMA).
// Q/Kt read from hp ([node][d], d-contiguous); V from hpT ([d][node]).
__global__ __launch_bounds__(128)
void flash_gat_kernel(const bf16_t* __restrict__ hp, const bf16_t* __restrict__ hpT,
                      float* __restrict__ gat) {
  __shared__ bf16_t pl[4][16][32];               // per-wave P tile
  int wave = threadIdx.x >> 5, lane = threadIdx.x & 31;
  int task = blockIdx.x * 4 + wave;              // B*H*(N/16) = 2048 tasks, exact grid
  int b    = task >> 9;
  int h    = (task >> 7) & 3;
  int tile = task & 127;
  int g = lane >> 4, r = lane & 15;

  const bf16_t* base  = hp  + (long)b * 2048 * 256 + h * 64;
  const bf16_t* vbase = hpT + (long)(b * 4 + h) * 64 * 2048;

  AFrag qa[2];
  const bf16_t* qrow = base + (long)(tile * 16 + r) * 256;
#pragma unroll
  for (int kc = 0; kc < 2; ++kc) {
    qa[kc].q[0] = *(const u32x4*)(qrow + kc * 32 + g * 16);
    qa[kc].q[1] = *(const u32x4*)(qrow + kc * 32 + g * 16 + 8);
  }

  float m[8], l[8];
  v8f zero = {};
  v8f acc[4] = {zero, zero, zero, zero};
#pragma unroll
  for (int v = 0; v < 8; ++v) { m[v] = -3.0e38f; l[v] = 0.0f; }

  for (int j0 = 0; j0 < 2048; j0 += 32) {
    v8f s[2];
#pragma unroll
    for (int t = 0; t < 2; ++t) {
      v8f st = {};
#pragma unroll
      for (int kc = 0; kc < 2; ++kc) {
        BFrag kb;  // col = key node, 16 contiguous d per lane
        const u32x4* kp = (const u32x4*)(base + (long)(j0 + t * 16 + r) * 256 + kc * 32 + g * 16);
        kb.q[0] = kp[0];
        kb.q[1] = kp[1];
        st = WMMA_BF16(qa[kc].v, kb.v, st);
      }
#pragma unroll
      for (int v = 0; v < 8; ++v) {              // LeakyReLU(0.2)
        float x = st[v];
        st[v] = x >= 0.0f ? x : 0.2f * x;
      }
      s[t] = st;
    }

    // online softmax update (rows live in VGPR index v within each 16-lane half)
#pragma unroll
    for (int v = 0; v < 8; ++v) {
      float mx = fmaxf(s[0][v], s[1][v]);
#pragma unroll
      for (int off = 1; off < 16; off <<= 1) mx = fmaxf(mx, __shfl_xor(mx, off, 32));
      float mn = fmaxf(m[v], mx);
      float corr = __expf(m[v] - mn);
      float p0 = __expf(s[0][v] - mn);
      float p1 = __expf(s[1][v] - mn);
      s[0][v] = p0; s[1][v] = p1;
      float rs = p0 + p1;
#pragma unroll
      for (int off = 1; off < 16; off <<= 1) rs += __shfl_xor(rs, off, 32);
      l[v] = l[v] * corr + rs;
      m[v] = mn;
#pragma unroll
      for (int c = 0; c < 4; ++c) acc[c][v] *= corr;
    }

    // C-layout -> A-fragment reshape of P through per-wave LDS
#pragma unroll
    for (int t = 0; t < 2; ++t)
#pragma unroll
      for (int v = 0; v < 8; ++v)
        pl[wave][v + 8 * g][t * 16 + r] = (bf16_t)s[t][v];
    asm volatile("s_wait_dscnt 0" ::: "memory");

    AFrag pa;
    pa.q[0] = *(const u32x4*)&pl[wave][r][g * 16];
    pa.q[1] = *(const u32x4*)&pl[wave][r][g * 16 + 8];

#pragma unroll
    for (int c = 0; c < 4; ++c) {
      BFrag vb;  // col = d column, 16 contiguous keys per lane (from hpT)
      const u32x4* vp = (const u32x4*)(vbase + (long)(c * 16 + r) * 2048 + j0 + g * 16);
      vb.q[0] = vp[0];
      vb.q[1] = vp[1];
      acc[c] = WMMA_BF16(pa.v, vb.v, acc[c]);
    }
  }

#pragma unroll
  for (int v = 0; v < 8; ++v) {
    float inv = 1.0f / l[v];
    int row = tile * 16 + v + 8 * g;
#pragma unroll
    for (int c = 0; c < 4; ++c)
      gat[((long)b * 2048 + row) * 256 + h * 64 + c * 16 + r] = acc[c][v] * inv;
  }
}

// ================================================================ launcher
extern "C" void kernel_launch(void* const* d_in, const int* in_sizes, int n_in,
                              void* d_out, int out_size, void* d_ws, size_t ws_size,
                              hipStream_t stream) {
  (void)in_sizes; (void)n_in; (void)out_size; (void)ws_size;
  const float* x     = (const float*)d_in[0];
  const float* adj   = (const float*)d_in[1];
  const float* Ws    = (const float*)d_in[2];
  const float* bsage = (const float*)d_in[3];
  const float* g1    = (const float*)d_in[4];
  const float* b1    = (const float*)d_in[5];
  const float* Wh    = (const float*)d_in[6];
  const float* a_gat = (const float*)d_in[7];
  const float* cw    = (const float*)d_in[8];
  const float* cb    = (const float*)d_in[9];
  const float* g2    = (const float*)d_in[10];
  const float* b2    = (const float*)d_in[11];
  const float* a_tcn = (const float*)d_in[12];
  float* out = (float*)d_out;

  const long BN = 8192;       // B*N rows
  const long ACT = BN * 256;  // activation element count

  char* p = (char*)d_ws;
  auto alloc = [&](size_t bytes) -> char* {
    char* ret = p;
    p += (bytes + 255) & ~(size_t)255;
    return ret;
  };
  bf16_t* x_bf   = (bf16_t*)alloc(ACT * 2);
  bf16_t* WsT    = (bf16_t*)alloc(256l * 256 * 2);
  bf16_t* adj_bf = (bf16_t*)alloc(2048l * 2048 * 2);
  bf16_t* WhT    = (bf16_t*)alloc(4l * 64 * 256 * 2);
  bf16_t* wpackT = (bf16_t*)alloc(256l * 768 * 2);
  bf16_t* supT   = (bf16_t*)alloc(ACT * 2);      // [n=256][b*2048+node]
  float*  hbuf   = (float*)alloc(ACT * 4);
  float*  hbn    = (float*)alloc(ACT * 4);
  bf16_t* hbn_bf = (bf16_t*)alloc(ACT * 2);
  bf16_t* hp_bf  = (bf16_t*)alloc(ACT * 2);
  bf16_t* hpT    = (bf16_t*)alloc(ACT * 2);      // [b][h][d][node]
  float*  gat    = (float*)alloc(ACT * 4);
  float*  gf     = (float*)alloc(ACT * 4);
  bf16_t* g_bf   = (bf16_t*)alloc(ACT * 2);
  float*  tbuf   = (float*)alloc(ACT * 4);
  float*  mean1  = (float*)alloc(256 * 4);
  float*  rstd1  = (float*)alloc(256 * 4);
  float*  mean2  = (float*)alloc(256 * 4);
  float*  rstd2  = (float*)alloc(256 * 4);

  f2bf_kernel<<<(unsigned)((ACT + 255) / 256), 256, 0, stream>>>(x, x_bf, ACT);
  f2bf_kernel<<<(unsigned)((2048l * 2048 + 255) / 256), 256, 0, stream>>>(adj, adj_bf, 2048l * 2048);
  ws_transpose_kernel<<<256, 256, 0, stream>>>(Ws, WsT);
  wh_transpose_kernel<<<256, 256, 0, stream>>>(Wh, WhT);
  conv_pack_kernel<<<768, 256, 0, stream>>>(cw, wpackT);

  // supT = (x @ W_sage)^T   (transposed bf16 out only)
  gemm_tdm_kernel<0, false, false, true><<<dim3(4, 64, 1), 256, 0, stream>>>(
      x_bf, WsT, nullptr, nullptr, nullptr, supT,
      8192, 256, 256, 256, 256, 0, 8192, 0, 0, 0, 0);

  // h = relu(adj @ support + b_sage)   (per batch; BT = supT + z*2048)
  gemm_tdm_kernel<1, true, false, false><<<dim3(4, 16, 4), 256, 0, stream>>>(
      adj_bf, supT, bsage, hbuf, nullptr, nullptr,
      2048, 256, 2048, 2048, 8192, 256, 0, 0, 2048, 2048l * 256, 0);

  // bn1 + relu
  bn_stats_kernel<<<256, 256, 0, stream>>>(hbuf, (int)BN, 256, mean1, rstd1);
  bn_apply_relu_kernel<<<(unsigned)(ACT / 256), 256, 0, stream>>>(
      hbuf, mean1, rstd1, g1, b1, hbn, hbn_bf, ACT, 256);

  // hp[b, n, h*64:(h+1)*64] = h_bn @ Wh[h]
  gemm_tdm_kernel<0, false, true, false><<<dim3(1, 64, 4), 256, 0, stream>>>(
      hbn_bf, WhT, nullptr, nullptr, hp_bf, nullptr,
      8192, 64, 256, 256, 256, 256, 0, 0, 64l * 256, 64, 0);

  // hpT for the V operand
  hp_transpose_kernel<<<(unsigned)(ACT / 256), 256, 0, stream>>>(hp_bf, hpT);

  // flash attention: 2048 wave-tasks, 4 waves/block
  flash_gat_kernel<<<512, 128, 0, stream>>>(hp_bf, hpT, gat);

  // g = alpha_gat * gat + (1-alpha_gat) * h_bn
  mix_kernel<<<(unsigned)(ACT / 256), 256, 0, stream>>>(gat, hbn, a_gat, gf, g_bf, ACT);

  // t = conv1d(g) + conv_b  (K = 3*256 WMMA GEMM, zero-padded)
  wmma_conv_kernel<<<dim3(4, 16, 4), 256, 0, stream>>>(g_bf, wpackT, cb, tbuf);

  // bn2, relu, residual -> out
  bn_stats_kernel<<<256, 256, 0, stream>>>(tbuf, (int)BN, 256, mean2, rstd2);
  final_kernel<<<(unsigned)(ACT / 256), 256, 0, stream>>>(
      tbuf, mean2, rstd2, g2, b2, a_tcn, gf, out, ACT, 256);
}
